// GraphCare_32117765439836
// MI455X (gfx1250) — matrix operand order repaired and checked by reading, not compiled, not run on hardware
//
#include <hip/hip_runtime.h>

typedef __attribute__((ext_vector_type(16))) _Float16 v16h;
typedef __attribute__((ext_vector_type(8)))  _Float16 v8h;
typedef __attribute__((ext_vector_type(8)))  float    v8f;

#define N_NODES 50000
#define N_EDGES 800000
#define GNODES  4000
#define NRELS   200
#define HID     128
#define BATCHSZ 32
#define VIS     20
#define LAYERS  3
#define OUTC    100
#define DECAYF  0.03f

// ---------------------------------------------------------------------------
// WMMA fragment loaders (CDNA5 f16 16x16x32 layouts, CDNA5 ISA)
// A (16-bit 16x32): lanes 0-15 hold K{0..7,16..23}, lanes 16-31 K{8..15,24..31}
// B (32x16):        lanes 0-15 hold K{0..7,16..23}, lanes 16-31 K{8..15,24..31}
// ---------------------------------------------------------------------------
__device__ __forceinline__ v16h frag_a(const _Float16* rowp, int H) {
  v8h lo = *(const v8h*)(rowp + 8 * H);
  v8h hi = *(const v8h*)(rowp + 16 + 8 * H);
  return __builtin_shufflevector(lo, hi, 0,1,2,3,4,5,6,7,8,9,10,11,12,13,14,15);
}
__device__ __forceinline__ v16h frag_b(const _Float16* rowp, int H) {
  v8h lo = *(const v8h*)(rowp + 8 * H);
  v8h hi = *(const v8h*)(rowp + 16 + 8 * H);
  return __builtin_shufflevector(lo, hi, 0,1,2,3,4,5,6,7,8,9,10,11,12,13,14,15);
}

// ---------------------------------------------------------------------------
// Generic tiled WMMA GEMM: C[M,Nc] = A[M,K] * W[Nc,K]^T (+bias) (+relu)
// A is f32 (a_f16=0, converted while staging) or pre-converted f16 (a_f16=1).
// 256 thr = 8 waves, block tile 64x128, K-step 64 (zero-filled past K),
// 8 WMMAs per wave per barrier pair, f32 accumulate.
// ---------------------------------------------------------------------------
__global__ __launch_bounds__(256) void gemm_wmma(
    const void* __restrict__ Av, int a_f16, const float* __restrict__ W,
    const float* __restrict__ bias, float* __restrict__ C,
    int M, int K, int Nc, int relu)
{
  __shared__ __align__(16) _Float16 As[64][64];    // 8 KB
  __shared__ __align__(16) _Float16 Bs[128][64];   // 16 KB
  const int tid  = threadIdx.x;
  const int lane = tid & 31;
  const int wid  = tid >> 5;
  const int wm   = wid >> 2;     // 0..1 -> m sub-block
  const int wn   = wid & 3;      // 0..3 -> n sub-block
  const int H    = lane >> 4;    // lane half
  const int ln   = lane & 15;
  const int rowBase = blockIdx.y * 64;
  const int colBase = blockIdx.x * 128;

  v8f acc00 = {}, acc01 = {}, acc10 = {}, acc11 = {};

  for (int k0 = 0; k0 < K; k0 += 64) {
    // prefetch next K-slab of the streamed W operand (global_prefetch_b8)
    if (k0 + 64 < K) {
      int pr = tid >> 1;
      int pgc = colBase + pr; if (pgc >= Nc) pgc = Nc - 1;
      __builtin_prefetch(W + (size_t)pgc * K + k0 + 64 + (size_t)(tid & 1) * 32, 0, 1);
    }
    __syncthreads();
    // ---- stage A tile 64x64 ----
    if (a_f16) {
      const _Float16* Ah = (const _Float16*)Av;
      for (int i = tid; i < 64 * 8; i += 256) {         // 8-half (16B) chunks
        int r = i >> 3, g = (i & 7) << 3;
        int gr = rowBase + r; if (gr >= M) gr = M - 1;
        v8h v = {};
        if (k0 + g < K) v = *(const v8h*)(Ah + (size_t)gr * K + k0 + g);
        *(v8h*)&As[r][g] = v;
      }
    } else {
      const float* Af = (const float*)Av;
      for (int i = tid; i < 64 * 16; i += 256) {        // float4 chunks + cvt
        int r = i >> 4, g = (i & 15) << 2;
        int gr = rowBase + r; if (gr >= M) gr = M - 1;
        float4 v = make_float4(0.f, 0.f, 0.f, 0.f);
        if (k0 + g < K) v = *(const float4*)(Af + (size_t)gr * K + k0 + g);
        _Float16* p = &As[r][g];
        p[0] = (_Float16)v.x; p[1] = (_Float16)v.y;
        p[2] = (_Float16)v.z; p[3] = (_Float16)v.w;
      }
    }
    // ---- stage B tile 128x64: Bs[n][k] = W[colBase+n][k0+k] ----
    for (int i = tid; i < 128 * 16; i += 256) {
      int r = i >> 4, g = (i & 15) << 2;
      int gc = colBase + r; if (gc >= Nc) gc = Nc - 1;
      float4 v = make_float4(0.f, 0.f, 0.f, 0.f);
      if (k0 + g < K) v = *(const float4*)(W + (size_t)gc * K + k0 + g);
      _Float16* p = &Bs[r][g];
      p[0] = (_Float16)v.x; p[1] = (_Float16)v.y;
      p[2] = (_Float16)v.z; p[3] = (_Float16)v.w;
    }
    __syncthreads();

#pragma unroll
    for (int s = 0; s < 2; s++) {
      v16h a0 = frag_a(&As[32 * wm + ln][32 * s], H);
      v16h a1 = frag_a(&As[32 * wm + 16 + ln][32 * s], H);
      v16h b0 = frag_b(&Bs[32 * wn + ln][32 * s], H);
      v16h b1 = frag_b(&Bs[32 * wn + 16 + ln][32 * s], H);
      acc00 = __builtin_amdgcn_wmma_f32_16x16x32_f16(false, a0, false, b0, (short)0, acc00, false, false);
      acc01 = __builtin_amdgcn_wmma_f32_16x16x32_f16(false, a0, false, b1, (short)0, acc01, false, false);
      acc10 = __builtin_amdgcn_wmma_f32_16x16x32_f16(false, a1, false, b0, (short)0, acc10, false, false);
      acc11 = __builtin_amdgcn_wmma_f32_16x16x32_f16(false, a1, false, b1, (short)0, acc11, false, false);
    }
  }

  // C/D layout: VGPR r -> M = r + 8*H, N = lane&15
#define STORE_TILE(accv, mi, ni) do {                                   \
    int gc = colBase + 32 * wn + 16 * (ni) + ln;                        \
    if (gc < Nc) {                                                      \
      float bv = bias ? bias[gc] : 0.0f;                                \
      int gr0 = rowBase + 32 * wm + 16 * (mi) + 8 * H;                  \
      for (int r = 0; r < 8; r++) {                                     \
        int gr = gr0 + r;                                               \
        if (gr < M) {                                                   \
          float vv = (accv)[r] + bv;                                    \
          if (relu) vv = fmaxf(vv, 0.0f);                               \
          C[(size_t)gr * Nc + gc] = vv;                                 \
        }                                                               \
      }                                                                 \
    }                                                                   \
  } while (0)

  STORE_TILE(acc00, 0, 0); STORE_TILE(acc01, 0, 1);
  STORE_TILE(acc10, 1, 0); STORE_TILE(acc11, 1, 1);
#undef STORE_TILE
}

// f32 -> f16 bulk convert (for the reused visit operand)
__global__ __launch_bounds__(256) void cvt_f16_kernel(
    const float* __restrict__ s, _Float16* __restrict__ d, int n4) {
  int t = blockIdx.x * 256 + threadIdx.x;
  if (t >= n4) return;
  float4 v = ((const float4*)s)[t];
  _Float16* p = d + (size_t)t * 4;
  p[0] = (_Float16)v.x; p[1] = (_Float16)v.y;
  p[2] = (_Float16)v.z; p[3] = (_Float16)v.w;
}

// wrel[l][r] = dot(relP[r], wr_w[l]) + wr_b[l]
__global__ void wrel_kernel(const float* __restrict__ relP,
                            const float* __restrict__ wr_w,
                            const float* __restrict__ wr_b,
                            float* __restrict__ wrel) {
  int t = blockIdx.x * blockDim.x + threadIdx.x;
  if (t >= LAYERS * NRELS) return;
  int l = t / NRELS, r = t - l * NRELS;
  float a = wr_b[l];
  for (int c = 0; c < HID; c++) a += relP[r * HID + c] * wr_w[l * HID + c];
  wrel[t] = a;
}

// x[i] = Pnb[node_ids[i]] + lin_b ; nidx[i] = batch[i]*GNODES + node_ids[i]
__global__ __launch_bounds__(256) void gather_kernel(
    const float* __restrict__ Pnb, const float* __restrict__ lin_b,
    const int* __restrict__ node_ids, const int* __restrict__ batch,
    float* __restrict__ x, int* __restrict__ nidx) {
  int t = blockIdx.x * 256 + threadIdx.x;
  int i = t >> 5; if (i >= N_NODES) return;
  int c = (t & 31) << 2;
  int nid = node_ids[i];
  float4 v = *(const float4*)(Pnb + (size_t)nid * HID + c);
  float4 b = *(const float4*)(lin_b + c);
  v.x += b.x; v.y += b.y; v.z += b.z; v.w += b.w;
  *(float4*)(x + (size_t)i * HID + c) = v;
  if ((t & 31) == 0) nidx[i] = batch[i] * GNODES + nid;
}

// beta[b,v] = tanh(dot(visit[b,v,:], beta_w[l]) + beta_b[l]) * exp(DECAY*(V-v))
__global__ __launch_bounds__(256) void beta_kernel(
    const float* __restrict__ visit, const float* __restrict__ bw,
    const float* __restrict__ bb, float* __restrict__ betaB) {
  __shared__ float sh[256];
  int row = blockIdx.x, tid = threadIdx.x;
  float a = 0.f;
  for (int n = tid; n < GNODES; n += 256) a += visit[(size_t)row * GNODES + n] * bw[n];
  sh[tid] = a; __syncthreads();
  for (int s = 128; s > 0; s >>= 1) { if (tid < s) sh[tid] += sh[tid + s]; __syncthreads(); }
  if (tid == 0) {
    int v = row % VIS;
    float lam = expf(DECAYF * (float)(VIS - v));
    betaB[row] = tanhf(sh[0] + bb[0]) * lam;
  }
}

// attn[b,m] = sum_v softmax_v(S[b,v,m]) * beta[b,v]
__global__ __launch_bounds__(256) void attn_kernel(
    const float* __restrict__ S, const float* __restrict__ betaB,
    float* __restrict__ attn) {
  int idx = blockIdx.x * 256 + threadIdx.x;
  if (idx >= BATCHSZ * GNODES) return;
  int b = idx / GNODES, m = idx - b * GNODES;
  float sv[VIS];
  float mx = -1e30f;
  for (int v = 0; v < VIS; v++) {
    sv[v] = S[((size_t)(b * VIS + v)) * GNODES + m];
    mx = fmaxf(mx, sv[v]);
  }
  float den = 0.f, num = 0.f;
  for (int v = 0; v < VIS; v++) {
    float e = expf(sv[v] - mx);
    den += e;
    num += e * betaB[b * VIS + v];
  }
  attn[idx] = num / den;
}

__global__ __launch_bounds__(256) void copy4_kernel(const float* __restrict__ s,
                                                    float* __restrict__ d, int n4) {
  int t = blockIdx.x * 256 + threadIdx.x;
  if (t < n4) ((float4*)d)[t] = ((const float4*)s)[t];
}

// fused message + relu + scatter-add (x/agg/relP L2-resident; 32 thr/edge)
__global__ __launch_bounds__(256) void edge_kernel(
    const int* __restrict__ src, const int* __restrict__ dst,
    const int* __restrict__ rel, const float* __restrict__ x,
    const float* __restrict__ relP, const float* __restrict__ wrel_l,
    const float* __restrict__ attn, const int* __restrict__ nidx,
    float* __restrict__ agg) {
  int t = blockIdx.x * 256 + threadIdx.x;
  int e = t >> 5; if (e >= N_EDGES) return;
  int c = (t & 31) << 2;
  int s = src[e], d = dst[e], r = rel[e];
  float ea = attn[nidx[s]];
  float wr = wrel_l[r];
  float4 xv = *(const float4*)(x + (size_t)s * HID + c);
  float4 rp = *(const float4*)(relP + r * HID + c);
  float m0 = fmaxf(xv.x * ea + wr * rp.x, 0.f);
  float m1 = fmaxf(xv.y * ea + wr * rp.y, 0.f);
  float m2 = fmaxf(xv.z * ea + wr * rp.z, 0.f);
  float m3 = fmaxf(xv.w * ea + wr * rp.w, 0.f);
  float* p = agg + (size_t)d * HID + c;
  atomicAdd(p + 0, m0); atomicAdd(p + 1, m1);
  atomicAdd(p + 2, m2); atomicAdd(p + 3, m3);
}

__global__ void zero_kernel(float* __restrict__ p, int n) {
  int t = blockIdx.x * blockDim.x + threadIdx.x;
  if (t < n) p[t] = 0.f;
}

__global__ __launch_bounds__(256) void pool_acc(
    const float* __restrict__ x, const int* __restrict__ batch,
    float* __restrict__ psum, float* __restrict__ pcnt) {
  int t = blockIdx.x * 256 + threadIdx.x;
  int i = t >> 5; if (i >= N_NODES) return;
  int c = (t & 31) << 2;
  int b = batch[i];
  float4 v = *(const float4*)(x + (size_t)i * HID + c);
  float* p = psum + b * HID + c;
  atomicAdd(p + 0, v.x); atomicAdd(p + 1, v.y);
  atomicAdd(p + 2, v.z); atomicAdd(p + 3, v.w);
  if ((t & 31) == 0) atomicAdd(pcnt + b, 1.0f);
}

__global__ __launch_bounds__(128) void pool_fin(
    const float* __restrict__ psum, const float* __restrict__ pcnt,
    float* __restrict__ xcat) {
  int b = blockIdx.x, c = threadIdx.x;
  float cnt = fmaxf(pcnt[b], 1.0f);
  xcat[b * 2 * HID + c] = psum[b * HID + c] / cnt;
}

__global__ __launch_bounds__(256) void rowsum_kernel(
    const float* __restrict__ ehr, float* __restrict__ rsum) {
  __shared__ float sh[256];
  int b = blockIdx.x, tid = threadIdx.x;
  float a = 0.f;
  for (int m = tid; m < GNODES; m += 256) a += ehr[(size_t)b * GNODES + m];
  sh[tid] = a; __syncthreads();
  for (int s = 128; s > 0; s >>= 1) { if (tid < s) sh[tid] += sh[tid + s]; __syncthreads(); }
  if (tid == 0) rsum[b] = sh[0];
}

// xcat[b][128+c] = (ehr[b] @ Pnb)[c] / rsum[b] + lin_b[c]
__global__ __launch_bounds__(128) void xnode_kernel(
    const float* __restrict__ ehr, const float* __restrict__ Pnb,
    const float* __restrict__ rsum, const float* __restrict__ lin_b,
    float* __restrict__ xcat) {
  __shared__ float es[128];
  int b = blockIdx.x, tid = threadIdx.x;
  float acc = 0.f;
  for (int mb = 0; mb < GNODES; mb += 128) {
    __syncthreads();
    int m = mb + tid;
    es[tid] = (m < GNODES) ? ehr[(size_t)b * GNODES + m] : 0.f;
    __syncthreads();
    int lim = min(128, GNODES - mb);
    for (int j = 0; j < lim; j++) acc += es[j] * Pnb[(size_t)(mb + j) * HID + tid];
  }
  xcat[b * 2 * HID + HID + tid] = acc / rsum[b] + lin_b[tid];
}

__global__ __launch_bounds__(128) void logits_kernel(
    const float* __restrict__ xcat, const float* __restrict__ mlp_w,
    const float* __restrict__ mlp_b, float* __restrict__ out) {
  __shared__ float xc[2 * HID];
  int b = blockIdx.x, tid = threadIdx.x;
  xc[tid] = xcat[b * 2 * HID + tid];
  xc[HID + tid] = xcat[b * 2 * HID + HID + tid];
  __syncthreads();
  if (tid < OUTC) {
    float a = mlp_b[tid];
    for (int k = 0; k < 2 * HID; k++) a += xc[k] * mlp_w[tid * 2 * HID + k];
    out[b * OUTC + tid] = a;
  }
}

// ---------------------------------------------------------------------------
extern "C" void kernel_launch(void* const* d_in, const int* in_sizes, int n_in,
                              void* d_out, int out_size, void* d_ws, size_t ws_size,
                              hipStream_t stream) {
  (void)in_sizes; (void)n_in; (void)out_size; (void)ws_size;
  const float* node_emb = (const float*)d_in[0];
  const float* rel_emb  = (const float*)d_in[1];
  const float* lin_w    = (const float*)d_in[2];
  const float* lin_b    = (const float*)d_in[3];
  const float* alpha_w  = (const float*)d_in[4];
  const float* alpha_b  = (const float*)d_in[5];
  const float* beta_w   = (const float*)d_in[6];
  const float* beta_b   = (const float*)d_in[7];
  const float* wr_w     = (const float*)d_in[8];
  const float* wr_b     = (const float*)d_in[9];
  const float* conv_w   = (const float*)d_in[10];
  const float* conv_b   = (const float*)d_in[11];
  const float* mlp_w    = (const float*)d_in[12];
  const float* mlp_b    = (const float*)d_in[13];
  const float* visit    = (const float*)d_in[14];
  const float* ehr      = (const float*)d_in[15];
  const int* node_ids   = (const int*)d_in[16];
  const int* rel_ids    = (const int*)d_in[17];
  const int* edge_index = (const int*)d_in[18];
  const int* batch      = (const int*)d_in[19];
  const int* src = edge_index;
  const int* dst = edge_index + N_EDGES;
  float* out = (float*)d_out;

  // workspace layout (~70 MB), 256B-aligned chunks
  char* ws = (char*)d_ws;
  size_t off = 0;
  auto alloc = [&](size_t bytes) -> char* {
    char* p = ws + off;
    off += (bytes + 255) & ~(size_t)255;
    return p;
  };
  float*    x      = (float*)alloc((size_t)N_NODES * HID * 4);          // 25.6 MB
  float*    agg    = (float*)alloc((size_t)N_NODES * HID * 4);          // 25.6 MB
  float*    S      = (float*)alloc((size_t)BATCHSZ * VIS * GNODES * 4); // 10.24 MB
  _Float16* visitH = (_Float16*)alloc((size_t)BATCHSZ * VIS * GNODES * 2); // 5.12 MB
  float*    Pnb    = (float*)alloc((size_t)GNODES * HID * 4);           // 2 MB
  float*    relP   = (float*)alloc((size_t)NRELS * HID * 4);
  float*    wrel   = (float*)alloc(LAYERS * NRELS * 4);
  float*    attn   = (float*)alloc(BATCHSZ * GNODES * 4);
  float*    betaB  = (float*)alloc(BATCHSZ * VIS * 4);
  int*      nidx   = (int*)alloc(N_NODES * 4);
  float*    xcat   = (float*)alloc(BATCHSZ * 2 * HID * 4);
  float*    psum   = (float*)alloc(BATCHSZ * HID * 4);
  float*    pcnt   = (float*)alloc(BATCHSZ * 4);
  float*    rsum   = (float*)alloc(BATCHSZ * 4);

  // --- one-time projections (unique rows, not per node-instance/edge) ---
  gemm_wmma<<<dim3(1, (GNODES + 63) / 64), 256, 0, stream>>>(
      node_emb, 0, lin_w, nullptr, Pnb, GNODES, HID, HID, 0);
  gemm_wmma<<<dim3(1, (NRELS + 63) / 64), 256, 0, stream>>>(
      rel_emb, 0, lin_w, lin_b, relP, NRELS, HID, HID, 0);
  wrel_kernel<<<(LAYERS * NRELS + 255) / 256, 256, 0, stream>>>(relP, wr_w, wr_b, wrel);
  gather_kernel<<<(N_NODES * 32 + 255) / 256, 256, 0, stream>>>(
      Pnb, lin_b, node_ids, batch, x, nidx);
  // visit reused 3 layers x 32 col-tiles -> convert to f16 once
  cvt_f16_kernel<<<(BATCHSZ * VIS * GNODES / 4 + 255) / 256, 256, 0, stream>>>(
      visit, visitH, BATCHSZ * VIS * GNODES / 4);

  for (int l = 0; l < LAYERS; l++) {
    beta_kernel<<<BATCHSZ * VIS, 256, 0, stream>>>(
        visit, beta_w + (size_t)l * GNODES, beta_b + l, betaB);
    // dominant GEMM: S[640,4000] = visit @ alpha_w[l]^T + alpha_b[l]
    gemm_wmma<<<dim3((GNODES + 127) / 128, (BATCHSZ * VIS + 63) / 64), 256, 0, stream>>>(
        visitH, 1, alpha_w + (size_t)l * GNODES * GNODES, alpha_b + (size_t)l * GNODES,
        S, BATCHSZ * VIS, GNODES, GNODES, 0);
    attn_kernel<<<(BATCHSZ * GNODES + 255) / 256, 256, 0, stream>>>(S, betaB, attn);
    // agg = x  (eps=0: out = agg + x)
    copy4_kernel<<<(N_NODES * HID / 4 + 255) / 256, 256, 0, stream>>>(
        x, agg, N_NODES * HID / 4);
    edge_kernel<<<(N_EDGES * 32 + 255) / 256, 256, 0, stream>>>(
        src, dst, rel_ids, x, relP, wrel + l * NRELS, attn, nidx, agg);
    // x = relu(agg @ conv_w[l]^T + conv_b[l])
    gemm_wmma<<<dim3(1, (N_NODES + 63) / 64), 256, 0, stream>>>(
        agg, 0, conv_w + (size_t)l * HID * HID, conv_b + (size_t)l * HID,
        x, N_NODES, HID, HID, 1);
  }

  // --- global mean pool + patient-node branch + MLP head ---
  zero_kernel<<<(BATCHSZ * HID + 255) / 256, 256, 0, stream>>>(psum, BATCHSZ * HID);
  zero_kernel<<<1, 64, 0, stream>>>(pcnt, BATCHSZ);
  pool_acc<<<(N_NODES * 32 + 255) / 256, 256, 0, stream>>>(x, batch, psum, pcnt);
  pool_fin<<<BATCHSZ, HID, 0, stream>>>(psum, pcnt, xcat);
  rowsum_kernel<<<BATCHSZ, 256, 0, stream>>>(ehr, rsum);
  xnode_kernel<<<BATCHSZ, HID, 0, stream>>>(ehr, Pnb, rsum, lin_b, xcat);
  logits_kernel<<<BATCHSZ, 128, 0, stream>>>(xcat, mlp_w, mlp_b, out);
}
